// NNEDI3FullImageFast_35124242547308
// MI455X (gfx1250) — compile-verified, matrix operand order checked by reading, not run on hardware
//
#include <hip/hip_runtime.h>
#include <hip/hip_bf16.h>

typedef __attribute__((ext_vector_type(16))) _Float16 v16h;
typedef __attribute__((ext_vector_type(8)))  float    v8f;

#define IMG_H 540
#define IMG_W 960
#define NNS   256
#define KPAD  64
#define KTAPS 48

__device__ __forceinline__ int reflect_idx(int i, int n) {
  i = (i < 0) ? -i : i;
  i = (i >= n) ? (2 * n - 2 - i) : i;
  return i;
}

// Convert conv_w1/conv_w2 (fp32 [256][48]) -> fused f16 weight matrix Wh[512][64],
// K padded 48->64 with zeros.  N = 0..255 -> bank1, 256..511 -> bank2.
__global__ void nnedi3_prep_weights(const float* __restrict__ w1,
                                    const float* __restrict__ w2,
                                    _Float16* __restrict__ Wh) {
  int idx = blockIdx.x * blockDim.x + threadIdx.x;
  if (idx >= 512 * KPAD) return;
  int n = idx / KPAD;
  int k = idx % KPAD;
  float v = 0.0f;
  if (k < KTAPS) v = (n < NNS) ? w1[n * KTAPS + k] : w2[(n - NNS) * KTAPS + k];
  Wh[idx] = (_Float16)v;
}

// One block per image row. 256 threads = 8 wave32; each wave owns 4 N-tiles
// (64 of the 512 fused channels). B fragments live in registers for the whole row.
__global__ void __launch_bounds__(256, 1)
nnedi3_main(const float* __restrict__ xin,
            const _Float16* __restrict__ Wh,
            const float* __restrict__ WS0,
            const float* __restrict__ WS1,
            float* __restrict__ out) {
  __shared__ float    sm[16][512];      // scaled conv outputs [pixel][channel]
  __shared__ float    win32[6][24];     // padded window (fp32, for stats)
  __shared__ _Float16 winh[6][24];      // padded window (f16, for A frags)
  __shared__ float    ldsMean[16], ldsM1[16], ldsM2[16];

  const int tid  = threadIdx.x;
  const int lane = tid & 31;
  const int wave = tid >> 5;
  const int half = (lane >> 4) & 1;     // 0: lanes 0-15, 1: lanes 16-31
  const int lm   = lane & 15;
  const int y    = blockIdx.x;

  // ---- Preload persistent B fragments + WS bias (reused across all 60 tiles) ----
  // 16-bit B 32x16 lane layout mirrors A: lane half selects K-offset +8,
  // VGPR v holds K pair (v<4 ? 2v : 8+2v).
  v16h  bfrag[4][2];
  float wsv[4];
#pragma unroll
  for (int t = 0; t < 4; ++t) {
    const int nt = wave * 4 + t;
    const int n  = nt * 16 + lm;
    wsv[t] = (n < NNS) ? WS0[n] : WS1[n - NNS];
#pragma unroll
    for (int ks = 0; ks < 2; ++ks) {
      v16h b;
#pragma unroll
      for (int v = 0; v < 8; ++v) {
        const int k = ((v < 4) ? 2 * v : 8 + 2 * v) + 8 * half + 32 * ks;
        b[2 * v]     = Wh[n * KPAD + k];
        b[2 * v + 1] = Wh[n * KPAD + k + 1];
      }
      bfrag[t][ks] = b;
    }
  }

  for (int xt = 0; xt < IMG_W / 16; ++xt) {
    const int x0 = xt * 16;
    __syncthreads();  // protect window/sm reuse across iterations

    // ---- Load 6x23 reflect-padded window (rows y-2..y+3, cols x0-3..x0+19) ----
    if (tid < 6 * 23) {
      const int r  = tid / 23;
      const int c  = tid % 23;
      const int gr = reflect_idx(y - 2 + r, IMG_H);
      const int gc = reflect_idx(x0 - 3 + c, IMG_W);
      const float v = xin[gr * IMG_W + gc];
      win32[r][c] = v;
      winh[r][c]  = (_Float16)v;
    }
    __syncthreads();

    // ---- Per-pixel mean / variance / 1/std ----
    if (tid < 16) {
      float s = 0.0f, s2 = 0.0f;
#pragma unroll
      for (int r = 0; r < 6; ++r)
#pragma unroll
        for (int c = 0; c < 8; ++c) {
          const float v = win32[r][tid + c];
          s += v;
          s2 += v * v;
        }
      const float mean = s * (1.0f / 48.0f);
      const float var  = s2 * (1.0f / 48.0f) - mean * mean;
      const bool  ok   = var >= 1.192092896e-07f;
      const float m2   = ok ? rsqrtf(var) : 0.0f;
      ldsMean[tid] = mean;
      ldsM2[tid]   = m2;
      ldsM1[tid]   = var * m2;
    }

    // ---- Build A fragments (16-bit A 16x32 layout): lane lm = pixel row M ----
    v16h afrag[2];
#pragma unroll
    for (int ks = 0; ks < 2; ++ks) {
      v16h a;
#pragma unroll
      for (int v = 0; v < 8; ++v) {
        const int k = ((v < 4) ? 2 * v : 8 + 2 * v) + 8 * half + 32 * ks;
        if (k < KTAPS) {
          const int kr = k >> 3, kc = k & 7;
          a[2 * v]     = winh[kr][lm + kc];
          a[2 * v + 1] = winh[kr][lm + kc + 1];
        } else {
          a[2 * v]     = (_Float16)0.0f;
          a[2 * v + 1] = (_Float16)0.0f;
        }
      }
      afrag[ks] = a;
    }
    __syncthreads();  // stats visible before scaling

    // ---- WMMA: 16 pixels x 64 channels per wave (K = 64 via 2 chained steps) ----
#pragma unroll
    for (int t = 0; t < 4; ++t) {
      const int nt = wave * 4 + t;
      v8f c = {};
      c = __builtin_amdgcn_wmma_f32_16x16x32_f16(false, afrag[0], false, bfrag[t][0],
                                                 (short)0, c, false, false);
      c = __builtin_amdgcn_wmma_f32_16x16x32_f16(false, afrag[1], false, bfrag[t][1],
                                                 (short)0, c, false, false);
      const int n = nt * 16 + lm;
#pragma unroll
      for (int r = 0; r < 8; ++r) {
        const int m = r + 8 * half;     // C layout: VGPR r -> M=r (+8 for hi half)
        sm[m][n] = c[r] * ldsM2[m] + wsv[t];
      }
    }
    __syncthreads();

    // ---- Softmax / softsign reduction: 16 threads per pixel, shfl_xor tree ----
    {
      const int m = tid >> 4;
      const int j = tid & 15;
      float mx = -3.4e38f;
#pragma unroll
      for (int t = 0; t < 16; ++t) mx = fmaxf(mx, sm[m][j + 16 * t]);
#pragma unroll
      for (int s = 8; s >= 1; s >>= 1) mx = fmaxf(mx, __shfl_xor(mx, s, 32));

      float wsum = 0.0f, vsum = 0.0f;
#pragma unroll
      for (int t = 0; t < 16; ++t) {
        const int n  = j + 16 * t;
        const float s1 = sm[m][n];
        const float s2 = sm[m][n + NNS];
        const float e  = __expf(s1 - mx);
        const float a2 = s2 / (1.0f + fabsf(s2));
        wsum += e;
        vsum += e * a2;
      }
#pragma unroll
      for (int s = 8; s >= 1; s >>= 1) {
        wsum += __shfl_xor(wsum, s, 32);
        vsum += __shfl_xor(vsum, s, 32);
      }

      if (j == 0) {
        const int xc = x0 + m;
        float interp = ldsMean[m] + 5.0f * vsum / wsum * ldsM1[m];
        interp = fminf(fmaxf(interp, 0.0f), 1.0f);
        out[(2 * y) * IMG_W + xc]     = xin[y * IMG_W + xc];  // original row
        out[(2 * y + 1) * IMG_W + xc] = interp;               // interpolated row
      }
    }
  }
}

extern "C" void kernel_launch(void* const* d_in, const int* in_sizes, int n_in,
                              void* d_out, int out_size, void* d_ws, size_t ws_size,
                              hipStream_t stream) {
  const float* x   = (const float*)d_in[0];   // [1,1,540,960]
  const float* w1  = (const float*)d_in[1];   // [256,1,6,8]
  const float* w2  = (const float*)d_in[2];   // [256,1,6,8]
  const float* ws0 = (const float*)d_in[3];   // [256]
  const float* ws1 = (const float*)d_in[4];   // [256]
  float* out = (float*)d_out;                 // [1,1,1080,960]

  _Float16* Wh = (_Float16*)d_ws;             // 512*64 f16 = 64 KB

  nnedi3_prep_weights<<<(512 * KPAD + 255) / 256, 256, 0, stream>>>(w1, w2, Wh);
  nnedi3_main<<<IMG_H, 256, 0, stream>>>(x, Wh, ws0, ws1, out);
}
